// BaseAttention_49452253446308
// MI455X (gfx1250) — compile-verified, matrix operand order checked by reading, not run on hardware
//
#include <hip/hip_runtime.h>

#define B_ 2
#define T_ 4096
#define C_ 512
#define H_ 8
#define HD_ 64
#define WIN_ 128
#define M_ (B_*T_)      /* 8192 */
#define N3_ (3*C_)      /* 1536 */

typedef __attribute__((ext_vector_type(16))) __bf16 v16bf;
typedef __attribute__((ext_vector_type(8)))  __bf16 v8bf;
typedef __attribute__((ext_vector_type(8)))  float  v8f;

__device__ __forceinline__ __bf16 to_bf(float f) { return (__bf16)f; }

// Build a 16-half A/B fragment from a row whose K values are contiguous:
// lane-group g needs K in {8g..8g+7} (halves 0..7) and {16+8g..23+8g} (8..15).
// Two 16-byte loads (ds_load_b128 / global_load_b128).
__device__ __forceinline__ v16bf frag_from_row(const __bf16* row, int g) {
  v8bf lo = *(const v8bf*)(row + 8*g);
  v8bf hi = *(const v8bf*)(row + 16 + 8*g);
  v16bf f;
  #pragma unroll
  for (int u = 0; u < 8; ++u) { f[u] = lo[u]; f[8+u] = hi[u]; }
  return f;
}

__device__ __forceinline__ float half16_max(float v) {
  v = fmaxf(v, __shfl_xor(v, 1));
  v = fmaxf(v, __shfl_xor(v, 2));
  v = fmaxf(v, __shfl_xor(v, 4));
  v = fmaxf(v, __shfl_xor(v, 8));
  return v;
}
__device__ __forceinline__ float half16_sum(float v) {
  v += __shfl_xor(v, 1);
  v += __shfl_xor(v, 2);
  v += __shfl_xor(v, 4);
  v += __shfl_xor(v, 8);
  return v;
}

// ---------------------------------------------------------------------------
// Kernel 1: qkv = x @ W_qkv.  Block tile 128x128, wave tile 32x64 (8 WMMA/K-step).
// Writes Q,K as [b][h][t][d] bf16 and V TRANSPOSED as [b][h][d][t] bf16.
// ---------------------------------------------------------------------------
__global__ __launch_bounds__(256) void qkv_gemm_kernel(
    const float* __restrict__ X, const float* __restrict__ W,
    __bf16* __restrict__ Qo, __bf16* __restrict__ Ko, __bf16* __restrict__ Vt)
{
  __shared__ __bf16 as_[128][40];   // [m][k], rows 80B (16B-aligned runs)
  __shared__ __bf16 wst[128][40];   // TRANSPOSED: [n][k]
  const int m0 = blockIdx.x * 128;
  const int n0 = blockIdx.y * 128;
  const int tid  = threadIdx.x;
  const int wave = tid >> 5, lane = tid & 31;
  const int wm = wave >> 1, wn = wave & 1;        // 4 x 2 wave grid
  const int lg = lane >> 4, ln = lane & 15;

  v8f acc[2][4] = {};

  for (int k0 = 0; k0 < C_; k0 += 32) {
    { // X tile 128x32 f32 -> bf16 (row-major)
      int r = tid >> 1, c = (tid & 1) * 16;
      const float* s = X + (size_t)(m0 + r) * C_ + k0 + c;
      #pragma unroll
      for (int u = 0; u < 16; ++u) as_[r][c + u] = to_bf(s[u]);
      const float* pf = (k0 + 32 < C_) ? s + 32 : s;
      __builtin_prefetch(pf, 0, 2);
    }
    { // W tile 32x128 f32 -> bf16, stored transposed [n][k]
      int r = tid >> 3, c = (tid & 7) * 16;
      const float* s = W + (size_t)(k0 + r) * N3_ + n0 + c;
      #pragma unroll
      for (int u = 0; u < 16; ++u) wst[c + u][r] = to_bf(s[u]);
      const float* pf = (k0 + 32 < C_) ? s + (size_t)32 * N3_ : s;
      __builtin_prefetch(pf, 0, 2);
    }
    __syncthreads();

    v16bf afr[2];
    #pragma unroll
    for (int sub = 0; sub < 2; ++sub)
      afr[sub] = frag_from_row(&as_[wm*32 + sub*16 + ln][0], lg);
    #pragma unroll
    for (int nn = 0; nn < 4; ++nn) {
      v16bf bfr = frag_from_row(&wst[wn*64 + nn*16 + ln][0], lg);
      #pragma unroll
      for (int sub = 0; sub < 2; ++sub)
        acc[sub][nn] = __builtin_amdgcn_wmma_f32_16x16x32_bf16(
            false, afr[sub], false, bfr, (short)0, acc[sub][nn], false, false);
    }
    __syncthreads();
  }

  // ---- epilogue: block-uniform routing, per-lane-constant base pointers ----
  const int which = n0 >> 9;               // 0=Q 1=K 2=V (uniform per block)
  const int b  = m0 >> 12;                 // uniform (128-row tiles never straddle b)
  const int t0 = (m0 & 4095) + wm*32 + 8*lg;
  const int nb = (n0 & 511) + wn*64;

  if (which < 2) {
    __bf16* dst = (which == 0) ? Qo : Ko;  // [b][h][t][d]
    #pragma unroll
    for (int nn = 0; nn < 4; ++nn) {
      int rem = nb + nn*16;                // uniform; 16-tile stays inside a head
      int h = rem >> 6, dbase = rem & 63;
      __bf16* base = dst + (((size_t)b * H_ + h) * T_) * HD_ + dbase + ln;
      #pragma unroll
      for (int sub = 0; sub < 2; ++sub)
        #pragma unroll
        for (int r = 0; r < 8; ++r)
          base[(size_t)(t0 + sub*16 + r) * HD_] = to_bf(acc[sub][nn][r]);
    }
  } else {                                  // V transposed: [b][h][d][t]
    #pragma unroll
    for (int nn = 0; nn < 4; ++nn) {
      int rem = nb + nn*16;
      int h = rem >> 6, dbase = rem & 63;
      __bf16* base = Vt + (((size_t)b * H_ + h) * HD_ + dbase + ln) * T_ + t0;
      #pragma unroll
      for (int sub = 0; sub < 2; ++sub)
        #pragma unroll
        for (int r = 0; r < 8; ++r)
          base[sub*16 + r] = to_bf(acc[sub][nn][r]);
    }
  }
}

// ---------------------------------------------------------------------------
// Kernel 2: banded flash attention; one wave per 16-query tile, 4 waves/block
// ---------------------------------------------------------------------------
__global__ __launch_bounds__(128) void band_attn_kernel(
    const __bf16* __restrict__ Qi, const __bf16* __restrict__ Ki,
    const __bf16* __restrict__ Vti, __bf16* __restrict__ Oo)
{
  __shared__ __bf16 pst[4][16][40];       // per-wave P staging (16 q x 32 keys)
  const int wave = threadIdx.x >> 5, lane = threadIdx.x & 31;
  const int lg = lane >> 4, ln = lane & 15;
  const int tile = blockIdx.x * 4 + wave;
  const int ntq = T_ / 16;
  const int qt = tile % ntq, bh = tile / ntq;
  const int i0 = qt * 16;
  const __bf16* Qp  = Qi  + (size_t)bh * T_ * HD_;
  const __bf16* Kp  = Ki  + (size_t)bh * T_ * HD_;
  const __bf16* Vtp = Vti + (size_t)bh * T_ * HD_;   // [d][t]
  const float scale = 0.125f;             // hd^-0.5

  // Q fragments: two K=32 chunks across hd=64 (contiguous 16B runs)
  const __bf16* qrow = Qp + (size_t)(i0 + ln) * HD_;
  v16bf aq[2];
  #pragma unroll
  for (int c = 0; c < 2; ++c) aq[c] = frag_from_row(qrow + c*32, lg);

  float mrow[8], lrow[8];
  v8f oacc[4] = {};
  #pragma unroll
  for (int r = 0; r < 8; ++r) { mrow[r] = -1e30f; lrow[r] = 0.f; }

  int jlo = i0 - WIN_; if (jlo < 0) jlo = 0; jlo &= ~31;
  int jhi = i0 + 16 + WIN_; if (jhi > T_) jhi = T_;

  for (int j0 = jlo; j0 < jhi; j0 += 32) {
    // ---- scores: Q @ K^T for 32 keys (two 16x16 tiles, K-dim hd=64) ----
    v8f s[2] = {};
    #pragma unroll
    for (int kt = 0; kt < 2; ++kt) {
      int j = j0 + kt*16 + ln;
      if (j >= T_) j = T_ - 1;            // masked below anyway
      const __bf16* krow = Kp + (size_t)j * HD_;
      #pragma unroll
      for (int c = 0; c < 2; ++c) {
        v16bf bk = frag_from_row(krow + c*32, lg);
        s[kt] = __builtin_amdgcn_wmma_f32_16x16x32_bf16(
            false, aq[c], false, bk, (short)0, s[kt], false, false);
      }
    }

    // ---- band mask + online softmax (row stats via 16-lane shuffles) ----
    #pragma unroll
    for (int r = 0; r < 8; ++r) {
      int i  = i0 + r + 8*lg;
      int ja = j0 + ln, jb = j0 + 16 + ln;
      bool oka = (ja < T_) && (ja - i <= WIN_) && (i - ja <= WIN_);
      bool okb = (jb < T_) && (jb - i <= WIN_) && (i - jb <= WIN_);
      float va = oka ? s[0][r] * scale : -1e30f;
      float vb = okb ? s[1][r] * scale : -1e30f;
      float mx = half16_max(fmaxf(va, vb));
      float mnew = fmaxf(mrow[r], mx);
      float alpha = __expf(mrow[r] - mnew);
      float pa = oka ? __expf(va - mnew) : 0.f;
      float pb = okb ? __expf(vb - mnew) : 0.f;
      float rsum = half16_sum(pa + pb);
      lrow[r] = lrow[r] * alpha + rsum;
      mrow[r] = mnew;
      #pragma unroll
      for (int nn = 0; nn < 4; ++nn) oacc[nn][r] *= alpha;
      pst[wave][r + 8*lg][ln]      = to_bf(pa);
      pst[wave][r + 8*lg][16 + ln] = to_bf(pb);
    }
    // same-wave LDS RAW: DS ops are in-order; make the wait explicit
    asm volatile("s_wait_dscnt 0" ::: "memory");

    // ---- P (A-layout from LDS) @ V (B-layout from transposed V) ----
    v16bf ap = frag_from_row(&pst[wave][ln][0], lg);
    #pragma unroll
    for (int nn = 0; nn < 4; ++nn) {
      const __bf16* vrow = Vtp + (size_t)(nn*16 + ln) * T_ + j0;
      v16bf bv = frag_from_row(vrow, lg);   // two global b128 loads
      oacc[nn] = __builtin_amdgcn_wmma_f32_16x16x32_bf16(
          false, ap, false, bv, (short)0, oacc[nn], false, false);
    }
  }

  // ---- normalize + write O as [b][t][h*64+d] bf16 ----
  int b = bh >> 3, h = bh & 7;
  __bf16* obase = Oo + ((size_t)(b * T_ + i0 + 8*lg)) * C_ + h * HD_ + ln;
  #pragma unroll
  for (int nn = 0; nn < 4; ++nn)
    #pragma unroll
    for (int r = 0; r < 8; ++r)
      obase[(size_t)r * C_ + nn*16] = to_bf(oacc[nn][r] / lrow[r]);
}

// ---------------------------------------------------------------------------
// Kernel 3: out = O @ W_out + b_out.  A is bf16 in workspace -> staged into
// LDS with CDNA5 async loads (global_load_async_to_lds_b128 + ASYNCcnt).
// ---------------------------------------------------------------------------
__global__ __launch_bounds__(256) void out_gemm_kernel(
    const __bf16* __restrict__ A, const float* __restrict__ W,
    const float* __restrict__ bias, float* __restrict__ Out)
{
  __shared__ __bf16 as_[128][40];
  __shared__ __bf16 wst[128][40];   // transposed [n][k]
  const int m0 = blockIdx.x * 128;
  const int n0 = blockIdx.y * 128;
  const int tid  = threadIdx.x;
  const int wave = tid >> 5, lane = tid & 31;
  const int wm = wave >> 1, wn = wave & 1;
  const int lg = lane >> 4, ln = lane & 15;

  v8f acc[2][4] = {};

  for (int k0 = 0; k0 < C_; k0 += 32) {
    { // A tile 128x32 bf16: async copy global -> LDS, 2 x 16B per thread
      int r = tid >> 1, c = (tid & 1) * 16;
      const __bf16* gp = A + (size_t)(m0 + r) * C_ + k0 + c;
      unsigned l0 = (unsigned)(uintptr_t)&as_[r][c];
      unsigned l1 = (unsigned)(uintptr_t)&as_[r][c + 8];
      asm volatile("global_load_async_to_lds_b128 %0, %1, off"
                   :: "v"(l0), "v"(gp) : "memory");
      asm volatile("global_load_async_to_lds_b128 %0, %1, off"
                   :: "v"(l1), "v"(gp + 8) : "memory");
    }
    { // W tile 32x128 f32 -> bf16, transposed
      int r = tid >> 3, c = (tid & 7) * 16;
      const float* s = W + (size_t)(k0 + r) * C_ + n0 + c;
      #pragma unroll
      for (int u = 0; u < 16; ++u) wst[c + u][r] = to_bf(s[u]);
      const float* pf = (k0 + 32 < C_) ? s + (size_t)32 * C_ : s;
      __builtin_prefetch(pf, 0, 2);
    }
    asm volatile("s_wait_asynccnt 0" ::: "memory");
    __syncthreads();

    v16bf afr[2];
    #pragma unroll
    for (int sub = 0; sub < 2; ++sub)
      afr[sub] = frag_from_row(&as_[wm*32 + sub*16 + ln][0], lg);
    #pragma unroll
    for (int nn = 0; nn < 4; ++nn) {
      v16bf bfr = frag_from_row(&wst[wn*64 + nn*16 + ln][0], lg);
      #pragma unroll
      for (int sub = 0; sub < 2; ++sub)
        acc[sub][nn] = __builtin_amdgcn_wmma_f32_16x16x32_bf16(
            false, afr[sub], false, bfr, (short)0, acc[sub][nn], false, false);
    }
    __syncthreads();
  }

  // ---- epilogue: hoist bias loads, per-lane-constant base pointer ----
  float bv[4];
  #pragma unroll
  for (int nn = 0; nn < 4; ++nn) bv[nn] = bias[n0 + wn*64 + nn*16 + ln];

  float* obase = Out + (size_t)(m0 + wm*32 + 8*lg) * C_ + n0 + wn*64 + ln;
  #pragma unroll
  for (int sub = 0; sub < 2; ++sub)
    #pragma unroll
    for (int r = 0; r < 8; ++r)
      #pragma unroll
      for (int nn = 0; nn < 4; ++nn)
        obase[(size_t)(sub*16 + r) * C_ + nn*16] = acc[sub][nn][r] + bv[nn];
}

// ---------------------------------------------------------------------------
extern "C" void kernel_launch(void* const* d_in, const int* in_sizes, int n_in,
                              void* d_out, int out_size, void* d_ws, size_t ws_size,
                              hipStream_t stream) {
  (void)in_sizes; (void)n_in; (void)out_size; (void)ws_size;
  const float* x    = (const float*)d_in[0];
  const float* Wqkv = (const float*)d_in[1];
  const float* Wout = (const float*)d_in[2];
  const float* bout = (const float*)d_in[3];
  float* out = (float*)d_out;

  // workspace: Q,K,Vt,O bf16 (each 8192*512 elements = 8.4 MB; 33.6 MB total)
  size_t qsz = (size_t)M_ * C_;
  __bf16* q  = (__bf16*)d_ws;
  __bf16* k  = q  + qsz;
  __bf16* vt = k  + qsz;
  __bf16* o  = vt + qsz;

  qkv_gemm_kernel<<<dim3(M_/128, N3_/128), 256, 0, stream>>>(x, Wqkv, q, k, vt);
  band_attn_kernel<<<dim3((B_*H_*(T_/16)) / 4), 128, 0, stream>>>(q, k, vt, o);
  out_gemm_kernel<<<dim3(M_/128, C_/128), 256, 0, stream>>>(o, Wout, bout, out);
}
